// TransformerDecoderLayer_34299608826000
// MI455X (gfx1250) — matrix operand order repaired and checked by reading, not compiled, not run on hardware
//
#include <hip/hip_runtime.h>
#include <hip/hip_bf16.h>

// ---------------------------------------------------------------------------
// MI455X (gfx1250) transformer decoder layer.
// All matmuls run on v_wmma_f32_16x16x32_bf16 (wave32, 16x16x32 tiles).
// Big intermediates (cross-attn K/V, 131072x512) kept in bf16 to halve HBM
// traffic (23.3 TB/s) and feed WMMA B-fragments directly from memory.
// Cross-attention uses single-pass online-softmax so the 100x16384 score
// matrix never hits HBM. Hot paths are specialized by template to remove
// per-iteration bounds/mask guards; tile staging is split into a batched
// load phase + convert/store phase for memory-level parallelism.
// ---------------------------------------------------------------------------

typedef __attribute__((ext_vector_type(16))) __bf16 v16bf;
typedef __attribute__((ext_vector_type(4)))  __bf16 v4bf;
typedef __attribute__((ext_vector_type(8)))  float  v8f;

union BF16x16 { v16bf v; __bf16 e[16]; };
union F32x8   { v8f   v; float  e[8];  };

#define NEG_INF (-__builtin_inff())

// ---------------------------------------------------------------------------
// Generic GEMM: C[M,N] = A[M,K] (fp32) @ W[N,K]^T (fp32) + bias, via bf16 WMMA.
// Block: 128 threads (4 waves), 64x64 C tile, K-step 32.
// MGUARD=false when M % 64 == 0 (no row-bound checks in the hot loop).
// ---------------------------------------------------------------------------
template <bool OUT_BF16, bool RELU, bool MGUARD>
__global__ __launch_bounds__(128) void gemm_wmma(
    const float* __restrict__ A, long lda,
    const float* __restrict__ W,              // N x K row-major
    const float* __restrict__ bias,
    void* __restrict__ C, long ldc,
    int M, int N, int K)
{
    __shared__ __bf16 smA[64 * 40];
    __shared__ __bf16 smB[64 * 40];

    const int tid  = threadIdx.x;
    const int lane = tid & 31;
    const int wave = tid >> 5;
    const int m0   = blockIdx.y * 64;
    const int n0   = blockIdx.x * 64;
    const int wr   = (wave >> 1) * 32;   // wave row offset in block tile
    const int wc   = (wave & 1) * 32;    // wave col offset in block tile
    const int l15  = lane & 15;
    const int halfk = (lane >= 16) ? 8 : 0;   // unused (A-frag also uses kb16)
    const int kb16  = (lane >= 16) ? 16 : 0;  // B-frag K base

    // staging coordinates (fixed per thread): quad idx = it*128 + tid
    int rA[4], cA[4];
    for (int it = 0; it < 4; ++it) {
        int idx = it * 128 + tid;
        rA[it] = idx >> 3;              // 0..63
        cA[it] = (idx & 7) << 2;        // 0,4,...,28
    }

    F32x8 acc[2][2];
    for (int i = 0; i < 2; ++i)
        for (int j = 0; j < 2; ++j)
            for (int r = 0; r < 8; ++r) acc[i][j].e[r] = 0.f;

    for (int k0 = 0; k0 < K; k0 += 32) {
        __syncthreads();
        // ---- load phase: batch all 8 global float4 loads ----
        float4 va[4], vb[4];
        for (int it = 0; it < 4; ++it) {
            int gr = m0 + rA[it];
            if (MGUARD && gr >= M) va[it] = make_float4(0.f, 0.f, 0.f, 0.f);
            else va[it] = *(const float4*)(A + (long)gr * lda + k0 + cA[it]);
        }
        for (int it = 0; it < 4; ++it)
            vb[it] = *(const float4*)(W + (long)(n0 + rA[it]) * K + k0 + cA[it]);
        // ---- convert + LDS store phase ----
        for (int it = 0; it < 4; ++it) {
            v4bf t;
            t[0] = (__bf16)va[it].x; t[1] = (__bf16)va[it].y;
            t[2] = (__bf16)va[it].z; t[3] = (__bf16)va[it].w;
            *(v4bf*)(&smA[rA[it] * 40 + cA[it]]) = t;
        }
        for (int it = 0; it < 4; ++it) {
            v4bf t;
            t[0] = (__bf16)vb[it].x; t[1] = (__bf16)vb[it].y;
            t[2] = (__bf16)vb[it].z; t[3] = (__bf16)vb[it].w;
            *(v4bf*)(&smB[rA[it] * 40 + cA[it]]) = t;
        }
        __syncthreads();

        BF16x16 af[2], bf[2];
        for (int i = 0; i < 2; ++i) {
            const __bf16* pa = &smA[(wr + i * 16 + l15) * 40];
            for (int j = 0; j < 16; ++j) af[i].e[j] = pa[kb16 + j];
            const __bf16* pb = &smB[(wc + i * 16 + l15) * 40 + kb16];
            for (int j = 0; j < 16; ++j) bf[i].e[j] = pb[j];
        }
        for (int i = 0; i < 2; ++i)
            for (int j = 0; j < 2; ++j)
                acc[i][j].v = __builtin_amdgcn_wmma_f32_16x16x32_bf16(
                    false, af[i].v, false, bf[j].v, (short)0, acc[i][j].v, false, false);
    }

    const int mo = (lane >= 16) ? 8 : 0;   // C layout: vgpr r -> row r+mo
    for (int i = 0; i < 2; ++i)
        for (int j = 0; j < 2; ++j)
            for (int r = 0; r < 8; ++r) {
                int gm = m0 + wr + i * 16 + mo + r;
                int gn = n0 + wc + j * 16 + l15;
                if (!MGUARD || gm < M) {
                    float v = acc[i][j].e[r] + bias[gn];
                    if (RELU) v = v > 0.f ? v : 0.f;
                    if (OUT_BF16) ((__bf16*)C)[(long)gm * ldc + gn] = (__bf16)v;
                    else          ((float*)C)[(long)gm * ldc + gn] = v;
                }
            }
}

// ---------------------------------------------------------------------------
// Flash attention, one wave (32 threads) per (b, h, 16-query tile).
// Q/K/V are bf16 with per-row strides; head dim = 32; online softmax.
// MASKED: apply bool mask (B*H, Sq, S), true = attend.
// KGUARD: S not a multiple of 32 -> guard key loads / mask tail keys.
// Out: fp32 (B, Sq, 256), heads concatenated.
// ---------------------------------------------------------------------------
template <bool MASKED, bool KGUARD>
__global__ __launch_bounds__(32) void flash_attn(
    const __bf16* __restrict__ Qb, int qStride,
    const __bf16* __restrict__ Kb, int kStride,
    const __bf16* __restrict__ Vb, int vStride,
    const unsigned char* __restrict__ mask,
    float* __restrict__ Out,
    int S, int Sq, float scale)
{
    __shared__ __bf16 smP[16 * 40];
    __shared__ __bf16 smV[32 * 40];

    const int lane = threadIdx.x;
    const int qt = blockIdx.x, h = blockIdx.y, b = blockIdx.z;
    const int q0 = qt * 16;
    const int l15   = lane & 15;
    const int halfk = (lane >= 16) ? 8 : 0;
    const int kb16  = (lane >= 16) ? 16 : 0;
    const int mo    = (lane >= 16) ? 8 : 0;

    // Q fragment (A layout): lane = q row, 32-deep K = head dim
    BF16x16 qf;
    {
        int qr = q0 + l15;
        if (qr < Sq) {
            const __bf16* p = Qb + ((long)b * Sq + qr) * qStride + h * 32;
            for (int i = 0; i < 16; ++i) qf.e[i] = p[kb16 + i];
        } else {
            for (int i = 0; i < 16; ++i) qf.e[i] = (__bf16)0.f;
        }
    }

    const unsigned char* mrow =
        MASKED ? mask + ((long)(b * 8 + h)) * (long)Sq * (long)S : nullptr;
    const __bf16* KbBH = Kb + (long)b * S * kStride + h * 32;
    const __bf16* VbBH = Vb + (long)b * S * vStride + h * 32;

    float mrun[8], lrun[8];
    F32x8 oacc[2];
    for (int j = 0; j < 8; ++j) { mrun[j] = NEG_INF; lrun[j] = 0.f; oacc[0].e[j] = 0.f; oacc[1].e[j] = 0.f; }

    const int nblk = (S + 31) & ~31;
    for (int k0 = 0; k0 < nblk; k0 += 32) {
        // scores: two 16x16 WMMAs (keys k0..k0+31)
        F32x8 sc[2];
        for (int nt = 0; nt < 2; ++nt) {
            BF16x16 kf;
            int key = k0 + nt * 16 + l15;
            if (!KGUARD || key < S) {
                const __bf16* p = KbBH + (long)key * kStride + kb16;
                for (int i = 0; i < 16; ++i) kf.e[i] = p[i];
            } else {
                for (int i = 0; i < 16; ++i) kf.e[i] = (__bf16)0.f;
            }
            for (int r = 0; r < 8; ++r) sc[nt].e[r] = 0.f;
            sc[nt].v = __builtin_amdgcn_wmma_f32_16x16x32_bf16(
                false, qf.v, false, kf.v, (short)0, sc[nt].v, false, false);
        }

        // stage V tile transposed into LDS: smV[dim][key]
        {
            int key = k0 + lane;
            if (!KGUARD || key < S) {
                const __bf16* p = VbBH + (long)key * vStride;
                for (int d = 0; d < 32; ++d) smV[d * 40 + lane] = p[d];
            } else {
                for (int d = 0; d < 32; ++d) smV[d * 40 + lane] = (__bf16)0.f;
            }
        }

        // mask + scale + online softmax (row stats across 16-lane groups)
        float alpha[8], pv0[8], pv1[8];
        int key0 = k0 + l15, key1 = k0 + 16 + l15;
        for (int j = 0; j < 8; ++j) {
            int qr = q0 + mo + j;
            bool okr = qr < Sq;
            bool a0 = okr && (!KGUARD || key0 < S) && (!MASKED || mrow[(long)qr * S + key0]);
            bool a1 = okr && (!KGUARD || key1 < S) && (!MASKED || mrow[(long)qr * S + key1]);
            float s0 = a0 ? sc[0].e[j] * scale : NEG_INF;
            float s1 = a1 ? sc[1].e[j] * scale : NEG_INF;
            float mx = fmaxf(s0, s1);
            for (int off = 1; off < 16; off <<= 1) mx = fmaxf(mx, __shfl_xor(mx, off, 32));
            float mnew = fmaxf(mrun[j], mx);
            float a = (mrun[j] == NEG_INF) ? 0.f : __expf(mrun[j] - mnew);
            float p0 = (s0 == NEG_INF) ? 0.f : __expf(s0 - mnew);
            float p1 = (s1 == NEG_INF) ? 0.f : __expf(s1 - mnew);
            float ps = p0 + p1;
            for (int off = 1; off < 16; off <<= 1) ps += __shfl_xor(ps, off, 32);
            mrun[j] = mnew;
            lrun[j] = lrun[j] * a + ps;
            alpha[j] = a;
            pv0[j] = p0; pv1[j] = p1;
        }

        // P (C layout) -> LDS row-major 16x32 -> reread as A fragment
        for (int j = 0; j < 8; ++j) {
            int row = mo + j;
            smP[row * 40 + l15]      = (__bf16)pv0[j];
            smP[row * 40 + 16 + l15] = (__bf16)pv1[j];
        }
        __syncthreads();   // single-wave workgroup: lowers to S_NOP, keeps LDS order

        BF16x16 pf;
        {
            const __bf16* p = &smP[l15 * 40];
            for (int i = 0; i < 16; ++i) pf.e[i] = p[kb16 + i];
        }

        // out += P @ V  (two dim tiles), with running rescale by alpha
        for (int dt = 0; dt < 2; ++dt) {
            BF16x16 vf;
            const __bf16* p = &smV[(dt * 16 + l15) * 40 + kb16];
            for (int i = 0; i < 16; ++i) vf.e[i] = p[i];
            for (int j = 0; j < 8; ++j) oacc[dt].e[j] *= alpha[j];
            oacc[dt].v = __builtin_amdgcn_wmma_f32_16x16x32_bf16(
                false, pf.v, false, vf.v, (short)0, oacc[dt].v, false, false);
        }
        __syncthreads();
    }

    for (int j = 0; j < 8; ++j) {
        int qr = q0 + mo + j;
        if (qr < Sq) {
            float inv = (lrun[j] > 0.f) ? 1.f / lrun[j] : 0.f;
            long base = ((long)b * Sq + qr) * 256 + h * 32;
            Out[base + l15]      = oacc[0].e[j] * inv;
            Out[base + 16 + l15] = oacc[1].e[j] * inv;
        }
    }
}

// ---------------------------------------------------------------------------
// Out = LayerNorm(X + Yadd) * g + b, one block (256 threads) per row, D=256.
// ---------------------------------------------------------------------------
__global__ __launch_bounds__(256) void add_ln(
    const float* __restrict__ X, const float* __restrict__ Yadd,
    const float* __restrict__ g, const float* __restrict__ be,
    float* __restrict__ Out)
{
    __shared__ float red[16];
    const int D = 256;
    const int row = blockIdx.x;
    const int t = threadIdx.x;
    float v = X[(long)row * D + t] + Yadd[(long)row * D + t];
    float s = v, s2 = v * v;
    for (int off = 1; off < 32; off <<= 1) {
        s  += __shfl_xor(s,  off, 32);
        s2 += __shfl_xor(s2, off, 32);
    }
    if ((t & 31) == 0) { red[t >> 5] = s; red[8 + (t >> 5)] = s2; }
    __syncthreads();
    float ts = 0.f, ts2 = 0.f;
    for (int i = 0; i < 8; ++i) { ts += red[i]; ts2 += red[8 + i]; }
    float mean = ts * (1.f / D);
    float var  = ts2 * (1.f / D) - mean * mean;
    float rstd = rsqrtf(var + 1e-5f);
    Out[(long)row * D + t] = (v - mean) * rstd * g[t] + be[t];
}

// ---------------------------------------------------------------------------
extern "C" void kernel_launch(void* const* d_in, const int* in_sizes, int n_in,
                              void* d_out, int out_size, void* d_ws, size_t ws_size,
                              hipStream_t stream)
{
    const int B = 8, Q = 100, D = 256, H = 8, HW = 128 * 128, FF = 2048;
    const int M = B * Q;                 // 800
    const int MP = B * HW;               // 131072
    const float scale = 0.17677669529663687f;   // 1/sqrt(32)

    const float* queries = (const float*)d_in[0];
    const float* pix     = (const float*)d_in[1];
    const unsigned char* mask = (const unsigned char*)d_in[2];
    const float* sa_in_w = (const float*)d_in[3];
    const float* sa_in_b = (const float*)d_in[4];
    const float* sa_out_w = (const float*)d_in[5];
    const float* sa_out_b = (const float*)d_in[6];
    const float* n1g = (const float*)d_in[7];
    const float* n1b = (const float*)d_in[8];
    const float* ca_in_w = (const float*)d_in[9];
    const float* ca_in_b = (const float*)d_in[10];
    const float* ca_out_w = (const float*)d_in[11];
    const float* ca_out_b = (const float*)d_in[12];
    const float* n2g = (const float*)d_in[13];
    const float* n2b = (const float*)d_in[14];
    const float* ff_w1 = (const float*)d_in[15];
    const float* ff_b1 = (const float*)d_in[16];
    const float* ff_w2 = (const float*)d_in[17];
    const float* ff_b2 = (const float*)d_in[18];
    const float* n3g = (const float*)d_in[19];
    const float* n3b = (const float*)d_in[20];

    // workspace carve-up (256B aligned)
    size_t off = 0;
    auto carve = [&](size_t bytes) {
        char* p = (char*)d_ws + off;
        off += (bytes + 255) & ~(size_t)255;
        return p;
    };
    __bf16* sa_qkv  = (__bf16*)carve((size_t)M * 3 * D * sizeof(__bf16));   // (800,768) bf16
    float*  sa_attn = (float*) carve((size_t)M * D * sizeof(float));
    float*  sa_proj = (float*) carve((size_t)M * D * sizeof(float));
    float*  x1      = (float*) carve((size_t)M * D * sizeof(float));
    __bf16* ca_q    = (__bf16*)carve((size_t)M * D * sizeof(__bf16));
    __bf16* ca_kv   = (__bf16*)carve((size_t)MP * 2 * D * sizeof(__bf16));  // (131072,512) bf16, ~134 MB
    float*  ca_attn = (float*) carve((size_t)M * D * sizeof(float));
    float*  ca_proj = (float*) carve((size_t)M * D * sizeof(float));
    float*  x2      = (float*) carve((size_t)M * D * sizeof(float));
    float*  ffn_h   = (float*) carve((size_t)M * FF * sizeof(float));
    float*  ffn_o   = (float*) carve((size_t)M * D * sizeof(float));
    (void)ws_size; (void)in_sizes; (void)n_in; (void)out_size;

    dim3 blk(128);
    const int mt = (M + 63) / 64;   // 13

    // 1) self-attn QKV: (800,256) @ (768,256)^T -> bf16 (800,768)
    gemm_wmma<true, false, true><<<dim3(3 * D / 64, mt), blk, 0, stream>>>(
        queries, D, sa_in_w, sa_in_b, sa_qkv, 3 * D, M, 3 * D, D);

    // 2) self-attention (S=100, no mask, key guard for tail)
    flash_attn<false, true><<<dim3(7, H, B), dim3(32), 0, stream>>>(
        sa_qkv + 0, 3 * D, sa_qkv + D, 3 * D, sa_qkv + 2 * D, 3 * D,
        nullptr, sa_attn, Q, Q, scale);

    // 3) self-attn out proj -> fp32
    gemm_wmma<false, false, true><<<dim3(D / 64, mt), blk, 0, stream>>>(
        sa_attn, D, sa_out_w, sa_out_b, sa_proj, D, M, D, D);

    // 4) x1 = LN(queries + sa_proj)
    add_ln<<<dim3(M), dim3(256), 0, stream>>>(queries, sa_proj, n1g, n1b, x1);

    // 5) cross-attn Q proj (rows 0..255 of ca_in_w) -> bf16
    gemm_wmma<true, false, true><<<dim3(D / 64, mt), blk, 0, stream>>>(
        x1, D, ca_in_w, ca_in_b, ca_q, D, M, D, D);

    // 6) cross-attn K/V proj of pixel features (rows 256..767) -> bf16 (131072,512)
    //    M = 131072 is a multiple of 64 -> no row guards in the hot loop.
    gemm_wmma<true, false, false><<<dim3(2 * D / 64, MP / 64), blk, 0, stream>>>(
        pix, D, ca_in_w + (size_t)D * D, ca_in_b + D, ca_kv, 2 * D, MP, 2 * D, D);

    // 7) masked cross-attention (S=16384, multiple of 32 -> no key guards)
    flash_attn<true, false><<<dim3(7, H, B), dim3(32), 0, stream>>>(
        ca_q, D, ca_kv + 0, 2 * D, ca_kv + D, 2 * D,
        mask, ca_attn, HW, Q, scale);

    // 8) cross-attn out proj
    gemm_wmma<false, false, true><<<dim3(D / 64, mt), blk, 0, stream>>>(
        ca_attn, D, ca_out_w, ca_out_b, ca_proj, D, M, D, D);

    // 9) x2 = LN(x1 + ca_proj)
    add_ln<<<dim3(M), dim3(256), 0, stream>>>(x1, ca_proj, n2g, n2b, x2);

    // 10) FFN up + ReLU: (800,256) -> (800,2048)
    gemm_wmma<false, true, true><<<dim3(FF / 64, mt), blk, 0, stream>>>(
        x2, D, ff_w1, ff_b1, ffn_h, FF, M, FF, D);

    // 11) FFN down: (800,2048) -> (800,256)
    gemm_wmma<false, false, true><<<dim3(D / 64, mt), blk, 0, stream>>>(
        ffn_h, FF, ff_w2, ff_b2, ffn_o, D, M, D, FF);

    // 12) out = LN(x2 + ffn_o)
    add_ln<<<dim3(M), dim3(256), 0, stream>>>(x2, ffn_o, n3g, n3b, (float*)d_out);
}
